// RboxSumGenericRoiExtractor_73718818668585
// MI455X (gfx1250) — compile-verified, matrix operand order checked by reading, not run on hardware
//
#include <hip/hip_runtime.h>

typedef __attribute__((ext_vector_type(16))) _Float16 v16h;
typedef __attribute__((ext_vector_type(8)))  _Float16 v8h;
typedef __attribute__((ext_vector_type(8)))  float    v8f;

#define C_CH   256
#define M_ROWS 49
#define M_PAD  64
#define K_TOT  2304      // 9 taps * 256 ci
#define K_STEP 32
#define K_ITERS (K_TOT / K_STEP)   // 72

// ---------------------------------------------------------------------------
// Kernel 1: reorder conv weights [co, ci, 3, 3] fp32 -> W^T [co][tap*256+ci] f16
// ---------------------------------------------------------------------------
__global__ void weight_prep_kernel(const float* __restrict__ wpre,
                                   const float* __restrict__ wpost,
                                   _Float16* __restrict__ tpre,
                                   _Float16* __restrict__ tpost) {
    int idx = blockIdx.x * blockDim.x + threadIdx.x;
    const int total = C_CH * K_TOT;
    if (idx >= 2 * total) return;
    const float* src = (idx < total) ? wpre : wpost;
    _Float16*    dst = (idx < total) ? tpre : tpost;
    int i   = (idx < total) ? idx : idx - total;
    int co  = i / K_TOT;
    int k   = i % K_TOT;
    int tap = k / C_CH;        // dy*3+dx
    int ci  = k % C_CH;
    dst[(size_t)co * K_TOT + k] =
        (_Float16)src[((size_t)co * C_CH + ci) * 9 + tap];
}

// ---------------------------------------------------------------------------
// Kernel 2: rotated RoIAlign (OUT=7, samples=2x2) -> NHWC f16 [N][49][256]
// one block per (roi, bin); 256 threads = channels
// ---------------------------------------------------------------------------
__global__ void __launch_bounds__(256)
roialign_rot_kernel(const float* __restrict__ feat,
                    const float* __restrict__ rois,
                    _Float16* __restrict__ out,
                    float scale, int H, int W) {
    int nb  = blockIdx.x;
    int n   = nb / 49;
    int bin = nb % 49;
    int ph  = bin / 7, pw = bin % 7;
    int c   = threadIdx.x;

    const float* r = rois + (size_t)n * 6;
    int   b  = (int)r[0];
    float cx = r[1] * scale, cy = r[2] * scale;
    float rw = r[3] * scale, rh = r[4] * scale;
    float th = r[5];
    float bw = rw * (1.0f / 7.0f), bh = rh * (1.0f / 7.0f);
    float ct = cosf(th), st = sinf(th);

    const float* fb = feat + ((size_t)(b * C_CH + c)) * H * W;
    float sum = 0.0f;
#pragma unroll
    for (int sy = 0; sy < 2; ++sy) {
#pragma unroll
        for (int sx = 0; sx < 2; ++sx) {
            float offy = (float)ph + ((float)sy + 0.5f) * 0.5f;
            float offx = (float)pw + ((float)sx + 0.5f) * 0.5f;
            float yy = -0.5f * rh + offy * bh;
            float xx = -0.5f * rw + offx * bw;
            float x = xx * ct + yy * st + cx;
            float y = yy * ct - xx * st + cy;
            bool valid = (y > -1.0f) && (y < (float)H) &&
                         (x > -1.0f) && (x < (float)W);
            y = fmaxf(y, 0.0f);
            x = fmaxf(x, 0.0f);
            int y0 = min((int)floorf(y), H - 1);
            int x0 = min((int)floorf(x), W - 1);
            int y1 = min(y0 + 1, H - 1);
            int x1 = min(x0 + 1, W - 1);
            if (y0 >= H - 1) y = (float)(H - 1);
            if (x0 >= W - 1) x = (float)(W - 1);
            float ly = y - (float)y0, lx = x - (float)x0;
            float hy = 1.0f - ly,     hx = 1.0f - lx;
            float v00 = fb[(size_t)y0 * W + x0];
            float v01 = fb[(size_t)y0 * W + x1];
            float v10 = fb[(size_t)y1 * W + x0];
            float v11 = fb[(size_t)y1 * W + x1];
            float v = hy * hx * v00 + hy * lx * v01 +
                      ly * hx * v10 + ly * lx * v11;
            sum += valid ? v : 0.0f;
        }
    }
    out[((size_t)n * 49 + bin) * C_CH + c] = (_Float16)(sum * 0.25f);
}

// ---------------------------------------------------------------------------
// Kernel 3: im2col 3x3 conv as WMMA GEMM. One block (8 waves) per RoI:
//   D[64(M) x 256(N)] = A[64 x 2304] * B[2304 x 256], K-step 32, f16->f32.
// Fragment packing follows ISA 16-bit A layout: lane&15 -> row,
// K-halves {0..7,16..23} / {8..15,24..31} selected by lane>>4.
// ---------------------------------------------------------------------------
template <bool INPUT_F32, bool STORE_NCHW>
__global__ void __launch_bounds__(256)
conv3x3_wmma_kernel(const void* __restrict__ in_,
                    const _Float16* __restrict__ wT,   // [256 co][2304 k] f16
                    const float* __restrict__ bias,
                    float* __restrict__ outp,
                    int beta) {
    __shared__ _Float16 Alds[M_PAD][K_STEP];   //  4 KB
    __shared__ _Float16 Blds[C_CH][K_STEP];    // 16 KB (stored as B^T tile)

    int n    = blockIdx.x;
    int tid  = threadIdx.x;
    int lane = tid & 31;
    int wave = tid >> 5;

    const _Float16* in16 = (const _Float16*)in_;
    const float*    in32 = (const float*)in_;

    v8f acc[2][4];
    const v8f zero8 = {0.f, 0.f, 0.f, 0.f, 0.f, 0.f, 0.f, 0.f};
#pragma unroll
    for (int j = 0; j < 2; ++j)
#pragma unroll
        for (int mt = 0; mt < 4; ++mt) acc[j][mt] = zero8;

    for (int kt = 0; kt < K_ITERS; ++kt) {
        int tap = kt >> 3;             // 0..8  (dy*3+dx)
        int cs  = (kt & 7) * 32;       // channel slice base
        int dy  = tap / 3 - 1;
        int dx  = tap % 3 - 1;

        // ---- cooperative im2col: A tile [64 rows x 32 k] ----
        {
            int m  = tid >> 2;            // 0..63
            int c8 = (tid & 3) * 8;       // 0,8,16,24
            int yy = m / 7 + dy;
            int xx = m % 7 + dx;
            bool ok = (m < M_ROWS) && (yy >= 0) && (yy < 7) &&
                      (xx >= 0) && (xx < 7);
            v8h a;
#pragma unroll
            for (int i = 0; i < 8; ++i) a[i] = (_Float16)0.0f;
            if (ok) {
                size_t base = (((size_t)n * 49 + yy * 7 + xx) * C_CH) + cs + c8;
                if (INPUT_F32) {
#pragma unroll
                    for (int i = 0; i < 8; ++i)
                        a[i] = (_Float16)in32[base + i];
                } else {
                    a = *(const v8h*)(in16 + base);
                }
            }
            *(v8h*)&Alds[m][c8] = a;
        }
        // ---- cooperative load: B^T tile [256 co x 32 k] ----
        {
            const _Float16* src = wT + (size_t)tid * K_TOT + (size_t)kt * K_STEP;
            *(v8h*)&Blds[tid][0]  = *(const v8h*)(src + 0);
            *(v8h*)&Blds[tid][8]  = *(const v8h*)(src + 8);
            *(v8h*)&Blds[tid][16] = *(const v8h*)(src + 16);
            *(v8h*)&Blds[tid][24] = *(const v8h*)(src + 24);
        }
        __syncthreads();

        int khalf = (lane >> 4) * 8;
        int rsel  = lane & 15;

        v16h bfrag[2];
#pragma unroll
        for (int j = 0; j < 2; ++j) {
            const _Float16* rp = &Blds[wave * 32 + j * 16 + rsel][0];
            v8h lo = *(const v8h*)(rp + khalf);
            v8h hi = *(const v8h*)(rp + khalf + 16);
#pragma unroll
            for (int i = 0; i < 8; ++i) { bfrag[j][i] = lo[i]; bfrag[j][i + 8] = hi[i]; }
        }
#pragma unroll
        for (int mt = 0; mt < 4; ++mt) {
            const _Float16* rp = &Alds[mt * 16 + rsel][0];
            v8h lo = *(const v8h*)(rp + khalf);
            v8h hi = *(const v8h*)(rp + khalf + 16);
            v16h afrag;
#pragma unroll
            for (int i = 0; i < 8; ++i) { afrag[i] = lo[i]; afrag[i + 8] = hi[i]; }
            acc[0][mt] = __builtin_amdgcn_wmma_f32_16x16x32_f16(
                false, afrag, false, bfrag[0], (short)0, acc[0][mt], false, false);
            acc[1][mt] = __builtin_amdgcn_wmma_f32_16x16x32_f16(
                false, afrag, false, bfrag[1], (short)0, acc[1][mt], false, false);
        }
        __syncthreads();
    }

    // ---- epilogue: bias + ReLU; accumulate NHWC fp32, or store NCHW fp32 ----
    int nsel  = lane & 15;
    int mhalf = (lane >> 4) * 8;
#pragma unroll
    for (int j = 0; j < 2; ++j) {
        int co = wave * 32 + j * 16 + nsel;
        float bv = bias[co];
#pragma unroll
        for (int mt = 0; mt < 4; ++mt) {
#pragma unroll
            for (int r = 0; r < 8; ++r) {
                int m = mt * 16 + mhalf + r;
                if (m < M_ROWS) {
                    float v = fmaxf(acc[j][mt][r] + bv, 0.0f);
                    if (STORE_NCHW) {
                        outp[((size_t)n * C_CH + co) * 49 + m] = v;
                    } else {
                        size_t idx = ((size_t)n * 49 + m) * C_CH + co;
                        float prev = beta ? outp[idx] : 0.0f;
                        outp[idx] = prev + v;
                    }
                }
            }
        }
    }
}

// ---------------------------------------------------------------------------
extern "C" void kernel_launch(void* const* d_in, const int* in_sizes, int n_in,
                              void* d_out, int out_size, void* d_ws, size_t ws_size,
                              hipStream_t stream) {
    const float* feat[4] = {(const float*)d_in[0], (const float*)d_in[1],
                            (const float*)d_in[2], (const float*)d_in[3]};
    const float* rois   = (const float*)d_in[4];
    const float* pre_w  = (const float*)d_in[5];
    const float* pre_b  = (const float*)d_in[6];
    const float* post_w = (const float*)d_in[7];
    const float* post_b = (const float*)d_in[8];
    float* outp = (float*)d_out;

    const int   N = 512;
    const int   HW[4]     = {256, 128, 64, 32};
    const float scales[4] = {0.25f, 0.125f, 0.0625f, 0.03125f};

    char* ws = (char*)d_ws;
    _Float16* roifeat = (_Float16*)ws;  ws += (size_t)N * 49 * C_CH * sizeof(_Float16);
    float*    accbuf  = (float*)ws;     ws += (size_t)N * 49 * C_CH * sizeof(float);
    _Float16* wpre    = (_Float16*)ws;  ws += (size_t)C_CH * K_TOT * sizeof(_Float16);
    _Float16* wpost   = (_Float16*)ws;

    {
        int total = 2 * C_CH * K_TOT;
        weight_prep_kernel<<<(total + 255) / 256, 256, 0, stream>>>(
            pre_w, post_w, wpre, wpost);
    }
    for (int l = 0; l < 4; ++l) {
        roialign_rot_kernel<<<N * 49, 256, 0, stream>>>(
            feat[l], rois, roifeat, scales[l], HW[l], HW[l]);
        conv3x3_wmma_kernel<false, false><<<N, 256, 0, stream>>>(
            (const void*)roifeat, wpre, pre_b, accbuf, l > 0 ? 1 : 0);
    }
    conv3x3_wmma_kernel<true, true><<<N, 256, 0, stream>>>(
        (const void*)accbuf, wpost, post_b, outp, 0);
}
